// GAE_49581102465576
// MI455X (gfx1250) — matrix operand order repaired and checked by reading, not compiled, not run on hardware
//
#include <hip/hip_runtime.h>
#include <hip/hip_bf16.h>

#define N_NODES 100000
#define N_EDGES 3200000
#define FEAT    256
#define HIDDEN  32
#define EMBED   16

typedef __attribute__((ext_vector_type(16))) _Float16 v16h;
typedef __attribute__((ext_vector_type(8)))  float    v8f;

// ---------- degree / normalization ----------
__global__ void deg_init(float* __restrict__ deg) {
    int n = blockIdx.x * blockDim.x + threadIdx.x;
    if (n < N_NODES) deg[n] = 1.0f;   // self-loop weight
}

__global__ void deg_scatter(const int* __restrict__ col, const float* __restrict__ ew,
                            float* __restrict__ deg) {
    int e = blockIdx.x * blockDim.x + threadIdx.x;
    if (e < N_EDGES) atomicAdd(&deg[col[e]], ew[e]);
}

__global__ void deg_rsqrt(float* __restrict__ deg) {
    int n = blockIdx.x * blockDim.x + threadIdx.x;
    if (n < N_NODES) deg[n] = rsqrtf(deg[n]);   // deg >= 1 always
}

// ---------- helpers ----------
__device__ __forceinline__ v16h pack16(float4 a, float4 b, float4 c, float4 d) {
    v16h r;
    r[0]=(_Float16)a.x;  r[1]=(_Float16)a.y;  r[2]=(_Float16)a.z;  r[3]=(_Float16)a.w;
    r[4]=(_Float16)b.x;  r[5]=(_Float16)b.y;  r[6]=(_Float16)b.z;  r[7]=(_Float16)b.w;
    r[8]=(_Float16)c.x;  r[9]=(_Float16)c.y;  r[10]=(_Float16)c.z; r[11]=(_Float16)c.w;
    r[12]=(_Float16)d.x; r[13]=(_Float16)d.y; r[14]=(_Float16)d.z; r[15]=(_Float16)d.w;
    return r;
}

__device__ __forceinline__ float4 relu4(float4 v) {
    v.x = fmaxf(v.x, 0.f); v.y = fmaxf(v.y, 0.f);
    v.z = fmaxf(v.z, 0.f); v.w = fmaxf(v.w, 0.f);
    return v;
}

// ---------- GEMM1: h1[N,32] = x[N,256] @ W1[32,256]^T via WMMA f16 ----------
// One wave per 16-node tile. A[m][k]=x[node][k]; B[k][n]=W1[n][k].
// CDNA5 16-bit A layout: lane m=L&15; lanes>=16 take K-offset +8; VGPR v holds
// K = (v<4?0:16)+khalf+2(v&3) .. +1  -> per-lane halves are 2x contiguous runs of 8.
// B layout: lane n=L&15; same K order as A (lanes>=16 take K-offset +8, two runs of 8).
__global__ __launch_bounds__(256) void gemm1_wmma(const float* __restrict__ x,
                                                  const float* __restrict__ W1,
                                                  float* __restrict__ h1) {
    int wave = threadIdx.x >> 5;
    int lane = threadIdx.x & 31;
    int tile = blockIdx.x * 8 + wave;
    if (tile * 16 >= N_NODES) return;          // wave-uniform guard (EXEC all-1s)
    int mbase = tile * 16;
    int m   = lane & 15;
    int hi  = lane >> 4;        // 0 or 1
    int khA = hi * 8;           // A K-offset within each 16-wide half
    int khB = hi * 16;          // unused; B loads use khA, same K order as A

    const float* xr  = x  + (size_t)(mbase + m) * FEAT;
    const float* w0r = W1 + (size_t)m        * FEAT;   // B columns 0..15
    const float* w1r = W1 + (size_t)(m + 16) * FEAT;   // B columns 16..31

    v8f acc0 = {}; v8f acc1 = {};
    #pragma unroll
    for (int kk = 0; kk < FEAT / 32; ++kk) {
        int kb = kk * 32;
        // A: 8 halves at K=kb+khA.. and 8 at K=kb+16+khA..
        v16h A = pack16(*(const float4*)(xr + kb + khA),
                        *(const float4*)(xr + kb + khA + 4),
                        *(const float4*)(xr + kb + 16 + khA),
                        *(const float4*)(xr + kb + 16 + khA + 4));
        // B: same K order as A from row n
        v16h B0 = pack16(*(const float4*)(w0r + kb + khA),
                         *(const float4*)(w0r + kb + khA + 4),
                         *(const float4*)(w0r + kb + 16 + khA),
                         *(const float4*)(w0r + kb + 16 + khA + 4));
        v16h B1 = pack16(*(const float4*)(w1r + kb + khA),
                         *(const float4*)(w1r + kb + khA + 4),
                         *(const float4*)(w1r + kb + 16 + khA),
                         *(const float4*)(w1r + kb + 16 + khA + 4));
        acc0 = __builtin_amdgcn_wmma_f32_16x16x32_f16(false, A, false, B0,
                                                      (short)0, acc0, false, false);
        acc1 = __builtin_amdgcn_wmma_f32_16x16x32_f16(false, A, false, B1,
                                                      (short)0, acc1, false, false);
    }
    // C layout: VGPR r -> row (r + hi*8), col = m
    int mrow = hi * 8;
    float* hb = h1 + (size_t)mbase * HIDDEN;
    #pragma unroll
    for (int r = 0; r < 8; ++r) {
        hb[(size_t)(r + mrow) * HIDDEN + m]      = acc0[r];
        hb[(size_t)(r + mrow) * HIDDEN + 16 + m] = acc1[r];
    }
}

// ---------- propagate init: out[n][f] = b[f] + h[n][f] * dis[n]^2 ----------
__global__ void prop_init(const float* __restrict__ h, const float* __restrict__ dis,
                          const float* __restrict__ b, float* __restrict__ out,
                          int shift /* log2(F) */) {
    int t = blockIdx.x * blockDim.x + threadIdx.x;
    int F = 1 << shift;
    if (t < N_NODES * F) {
        int n = t >> shift;
        int f = t & (F - 1);
        float d = dis[n];
        out[t] = b[f] + h[t] * d * d;
    }
}

// ---------- edge scatter, 32 features: 8 lanes per edge, 4 feats each ----------
__global__ void edge_scatter32(const int* __restrict__ row, const int* __restrict__ col,
                               const float* __restrict__ ew, const float* __restrict__ dis,
                               const float* __restrict__ h, float* __restrict__ out) {
    int t = blockIdx.x * blockDim.x + threadIdx.x;
    int e = t >> 3;
    if (e >= N_EDGES) return;
    int fo = (t & 7) * 4;
    int r = row[e], c = col[e];
    float w = dis[r] * ew[e] * dis[c];
    float4 hv = *(const float4*)(h + (size_t)r * HIDDEN + fo);
    float* ob = out + (size_t)c * HIDDEN + fo;
    atomicAdd(ob + 0, hv.x * w);
    atomicAdd(ob + 1, hv.y * w);
    atomicAdd(ob + 2, hv.z * w);
    atomicAdd(ob + 3, hv.w * w);
}

// ---------- GEMM2: z[N,16] = relu(out1)[N,32] @ W2[16,32]^T via WMMA ----------
__global__ __launch_bounds__(256) void gemm2_wmma(const float* __restrict__ out1,
                                                  const float* __restrict__ W2,
                                                  float* __restrict__ z) {
    int wave = threadIdx.x >> 5;
    int lane = threadIdx.x & 31;
    int tile = blockIdx.x * 8 + wave;
    if (tile * 16 >= N_NODES) return;
    int mbase = tile * 16;
    int m  = lane & 15;
    int hi = lane >> 4;
    int khA = hi * 8;
    int khB = hi * 16;

    const float* ar = out1 + (size_t)(mbase + m) * HIDDEN;
    v16h A = pack16(relu4(*(const float4*)(ar + khA)),
                    relu4(*(const float4*)(ar + khA + 4)),
                    relu4(*(const float4*)(ar + 16 + khA)),
                    relu4(*(const float4*)(ar + 16 + khA + 4)));
    const float* wr = W2 + (size_t)m * HIDDEN;   // B column n = m, K same order as A
    v16h B = pack16(*(const float4*)(wr + khA),
                    *(const float4*)(wr + khA + 4),
                    *(const float4*)(wr + 16 + khA),
                    *(const float4*)(wr + 16 + khA + 4));
    v8f acc = {};
    acc = __builtin_amdgcn_wmma_f32_16x16x32_f16(false, A, false, B,
                                                 (short)0, acc, false, false);
    int mrow = hi * 8;
    float* zb = z + (size_t)mbase * EMBED;
    #pragma unroll
    for (int r = 0; r < 8; ++r)
        zb[(size_t)(r + mrow) * EMBED + m] = acc[r];
}

// ---------- edge scatter, 16 features: 4 lanes per edge ----------
__global__ void edge_scatter16(const int* __restrict__ row, const int* __restrict__ col,
                               const float* __restrict__ ew, const float* __restrict__ dis,
                               const float* __restrict__ z, float* __restrict__ out) {
    int t = blockIdx.x * blockDim.x + threadIdx.x;
    int e = t >> 2;
    if (e >= N_EDGES) return;
    int fo = (t & 3) * 4;
    int r = row[e], c = col[e];
    float w = dis[r] * ew[e] * dis[c];
    float4 zv = *(const float4*)(z + (size_t)r * EMBED + fo);
    float* ob = out + (size_t)c * EMBED + fo;
    atomicAdd(ob + 0, zv.x * w);
    atomicAdd(ob + 1, zv.y * w);
    atomicAdd(ob + 2, zv.z * w);
    atomicAdd(ob + 3, zv.w * w);
}

extern "C" void kernel_launch(void* const* d_in, const int* in_sizes, int n_in,
                              void* d_out, int out_size, void* d_ws, size_t ws_size,
                              hipStream_t stream) {
    const float* x   = (const float*)d_in[0];
    const int*   ei  = (const int*)  d_in[1];
    const float* ew  = (const float*)d_in[2];
    const float* W1  = (const float*)d_in[3];
    const float* b1  = (const float*)d_in[4];
    const float* W2  = (const float*)d_in[5];
    const float* b2  = (const float*)d_in[6];
    float* out = (float*)d_out;

    const int* row = ei;             // ei[0]
    const int* col = ei + N_EDGES;   // ei[1]

    float* dis  = (float*)d_ws;                       // [N]
    float* h1   = dis  + N_NODES;                     // [N,32]
    float* out1 = h1   + (size_t)N_NODES * HIDDEN;    // [N,32]
    float* z    = out1 + (size_t)N_NODES * HIDDEN;    // [N,16]

    const int B = 256;
    // normalization
    deg_init   <<<(N_NODES + B - 1) / B, B, 0, stream>>>(dis);
    deg_scatter<<<(N_EDGES + B - 1) / B, B, 0, stream>>>(col, ew, dis);
    deg_rsqrt  <<<(N_NODES + B - 1) / B, B, 0, stream>>>(dis);

    // layer 1
    int tiles  = N_NODES / 16;                        // 6250, exact
    int gblk   = (tiles + 7) / 8;                     // 8 waves/block
    gemm1_wmma<<<gblk, B, 0, stream>>>(x, W1, h1);
    prop_init<<<((N_NODES * HIDDEN) + B - 1) / B, B, 0, stream>>>(h1, dis, b1, out1, 5);
    edge_scatter32<<<((size_t)N_EDGES * 8 + B - 1) / B, B, 0, stream>>>(row, col, ew, dis, h1, out1);

    // layer 2 (ReLU fused into GEMM2 A-load)
    gemm2_wmma<<<gblk, B, 0, stream>>>(out1, W2, z);
    prop_init<<<((N_NODES * EMBED) + B - 1) / B, B, 0, stream>>>(z, dis, b2, out, 4);
    edge_scatter16<<<((size_t)N_EDGES * 4 + B - 1) / B, B, 0, stream>>>(row, col, ew, dis, z, out);
}
